// FalconMambaMixer_74457553043829
// MI455X (gfx1250) — matrix operand order repaired and checked by reading, not compiled, not run on hardware
//
#include <hip/hip_runtime.h>
#include <hip/hip_bf16.h>

#define SEQ    1024
#define HID    2048
#define INTER  4096
#define NSTATE 16
#define DTRANK 128
#define KCONV  4
#define EPS    1e-6f

typedef __attribute__((ext_vector_type(16))) __bf16 bf16x16;
typedef __attribute__((ext_vector_type(8)))  __bf16 bf16x8;
typedef __attribute__((ext_vector_type(4)))  __bf16 bf16x4;
typedef __attribute__((ext_vector_type(8)))  float  f32x8;
typedef __attribute__((ext_vector_type(4)))  unsigned int u32x4;
typedef __attribute__((ext_vector_type(8)))  int i32x8;
typedef __attribute__((ext_vector_type(4)))  int i32x4;

union FragU { bf16x16 v16; bf16x8 v8[2]; };

#if defined(__has_builtin)
#if __has_builtin(__builtin_amdgcn_tensor_load_to_lds) && __has_builtin(__builtin_amdgcn_s_wait_tensorcnt)
#define HAVE_TDM 1
#endif
#endif

#ifdef HAVE_TDM
// ---------------------------------------------------------------------------
// 1-D f32 TDM copy: global -> LDS via Tensor Data Mover.
// D# group0: count=1 | lds_addr | global_addr(57b) | type=2
// D# group1: data_size=4B, tensor_dim0=nelem, tensor_dim1=1, tile_dim0=nelem,
//            tile_dim1=1, stride0=nelem.  (nelem <= 65535)
// Issue from ONE wave only (TDM ignores EXEC, issues once per wave).
// ---------------------------------------------------------------------------
__device__ __forceinline__ void tdm_copy_f32(void* lds_dst, const float* gsrc,
                                             unsigned nelem) {
  unsigned lds_off = (unsigned)(unsigned long long)lds_dst;  // addr[31:0] = LDS offset
  unsigned long long ga = (unsigned long long)gsrc;
  u32x4 g0;
  g0[0] = 1u;                                   // count=1, is_restore=0, gather=0
  g0[1] = lds_off;                              // lds_addr (bytes)
  g0[2] = (unsigned)(ga & 0xFFFFFFFFu);         // global_addr[31:0]
  g0[3] = (unsigned)((ga >> 32) & 0x1FFFFFFu) | (2u << 30);  // addr[56:32] | type=2
  i32x8 g1;
  g1[0] = (int)(2u << 16);                      // wg_mask=0, data_size=4B
  g1[1] = (int)((nelem & 0xFFFFu) << 16);       // abar=0, tensor_dim0[15:0]
  g1[2] = (int)(((nelem >> 16) & 0xFFFFu) | (1u << 16));  // td0[31:16], tensor_dim1=1
  g1[3] = (int)((nelem & 0xFFFFu) << 16);       // td1[31:16]=0, tile_dim0=nelem
  g1[4] = 1;                                    // tile_dim1=1, tile_dim2=0
  g1[5] = (int)nelem;                           // tensor_dim0_stride[31:0]
  g1[6] = 0;                                    // stride0[47:32]=0, stride1 lo=0
  g1[7] = 0;
  i32x4 z4 = {0, 0, 0, 0};
#if __clang_major__ >= 23
  i32x8 z8 = {0, 0, 0, 0, 0, 0, 0, 0};
  __builtin_amdgcn_tensor_load_to_lds(g0, g1, z4, z4, z8, 0);
#else
  __builtin_amdgcn_tensor_load_to_lds(g0, g1, z4, z4, 0);
#endif
}
#endif

// ---------------------------------------------------------------------------
// Generic GEMM: C[M,N] = A[M,K] * B[K,N], all f32 row-major in global memory.
// Tiles converted to bf16 in LDS, accumulated in f32 via v_wmma_f32_16x16x32_bf16.
// Requires M%128==0, N%128==0, K%32==0 (true for all call sites here).
// ---------------------------------------------------------------------------
#define BM 128
#define BN 128
#define BK 32

__global__ __launch_bounds__(256) void fm_gemm_bf16_wmma(
    const float* __restrict__ A, const float* __restrict__ B,
    float* __restrict__ C, int M, int N, int K) {
  __shared__ __align__(16) __bf16 sA[BM * BK];        // [m][k]
  __shared__ __align__(16) __bf16 sB[BN * BK];        // transposed: [n][k]

  const int tid  = threadIdx.x;
  const int m0   = blockIdx.y * BM;
  const int n0   = blockIdx.x * BN;
  const int wv   = tid >> 5;
  const int lane = tid & 31;
  const int half = lane >> 4;   // 0/1
  const int lm   = lane & 15;   // 0..15
  const int wr   = wv >> 2;     // 0..1  (wave row: 64 rows each)
  const int wc   = wv & 3;      // 0..3  (wave col: 32 cols each)

  f32x8 acc[4][2];
  #pragma unroll
  for (int a = 0; a < 4; ++a)
    #pragma unroll
    for (int b = 0; b < 2; ++b)
      #pragma unroll
      for (int e = 0; e < 8; ++e) acc[a][b][e] = 0.0f;

  for (int k0 = 0; k0 < K; k0 += BK) {
    // ---- stage A tile: 128x32 f32 -> bf16 LDS [m][k] ----
    #pragma unroll
    for (int j = 0; j < 4; ++j) {
      int q   = tid + 256 * j;      // 0..1023 float4s
      int row = q >> 3;             // 8 float4 per 32-wide row
      int cf  = q & 7;
      const float* gp = A + (size_t)(m0 + row) * K + k0 + cf * 4;
      float4 f = *(const float4*)gp;
      if (k0 + BK < K) __builtin_prefetch(gp + BK, 0, 3);
      bf16x4 p;
      p[0] = (__bf16)f.x; p[1] = (__bf16)f.y; p[2] = (__bf16)f.z; p[3] = (__bf16)f.w;
      *(bf16x4*)(sA + row * BK + cf * 4) = p;
    }
    // ---- stage B tile: 32x128 f32 -> bf16 LDS transposed [n][k] ----
    #pragma unroll
    for (int j = 0; j < 4; ++j) {
      int q   = tid + 256 * j;
      int row = q >> 5;             // k row 0..31 (32 float4 per 128-wide row)
      int cf  = q & 31;
      const float* gp = B + (size_t)(k0 + row) * N + n0 + cf * 4;
      float4 f = *(const float4*)gp;
      if (k0 + BK < K) __builtin_prefetch(gp + (size_t)BK * N, 0, 3);
      int c = cf * 4;
      sB[(c + 0) * BK + row] = (__bf16)f.x;
      sB[(c + 1) * BK + row] = (__bf16)f.y;
      sB[(c + 2) * BK + row] = (__bf16)f.z;
      sB[(c + 3) * BK + row] = (__bf16)f.w;
    }
    __syncthreads();

    // ---- load fragments per ISA 16-bit A(16x32) / B(32x16) lane layouts ----
    bf16x16 af[4], bf[2];
    #pragma unroll
    for (int mt = 0; mt < 4; ++mt) {
      const __bf16* base = sA + (wr * 64 + mt * 16 + lm) * BK;
      FragU u;
      u.v8[0] = *(const bf16x8*)(base + half * 8);        // K = half*8 .. +7
      u.v8[1] = *(const bf16x8*)(base + 16 + half * 8);   // K = 16+half*8 .. +7
      af[mt] = u.v16;
    }
    #pragma unroll
    for (int nt = 0; nt < 2; ++nt) {
      const __bf16* base = sB + (wc * 32 + nt * 16 + lm) * BK;
      FragU u;
      u.v8[0] = *(const bf16x8*)(base + half * 8);        // K = half*8 .. +7
      u.v8[1] = *(const bf16x8*)(base + 16 + half * 8);   // K = 16+half*8 .. +7
      bf[nt] = u.v16;
    }

    #pragma unroll
    for (int mt = 0; mt < 4; ++mt)
      #pragma unroll
      for (int nt = 0; nt < 2; ++nt)
        acc[mt][nt] = __builtin_amdgcn_wmma_f32_16x16x32_bf16(
            false, af[mt], false, bf[nt], (short)0, acc[mt][nt], false, false);
    __syncthreads();
  }

  // ---- epilogue: C/D layout -> lane (half, lm), VGPR v -> M = half*8+v ----
  #pragma unroll
  for (int mt = 0; mt < 4; ++mt)
    #pragma unroll
    for (int nt = 0; nt < 2; ++nt)
      #pragma unroll
      for (int v = 0; v < 8; ++v) {
        int row = m0 + wr * 64 + mt * 16 + half * 8 + v;
        int col = n0 + wc * 32 + nt * 16 + lm;
        C[(size_t)row * N + col] = acc[mt][nt][v];
      }
}

// ---------------------------------------------------------------------------
// Causal depthwise conv (K=4) over time + bias + SiLU.
// proj row stride 2*INTER; hidden = cols [0, INTER).
// ---------------------------------------------------------------------------
__global__ void fm_conv_silu(const float* __restrict__ proj,
                             const float* __restrict__ w,
                             const float* __restrict__ b,
                             float* __restrict__ hs) {
  int idx = blockIdx.x * blockDim.x + threadIdx.x;   // over SEQ*INTER
  int i = idx & (INTER - 1);
  int t = idx >> 12;
  float acc = b[i];
  #pragma unroll
  for (int k = 0; k < KCONV; ++k) {
    int tt = t - (KCONV - 1) + k;
    if (tt >= 0) acc += w[i * KCONV + k] * proj[(size_t)tt * (2 * INTER) + i];
  }
  hs[idx] = acc / (1.0f + __expf(-acc));
}

// ---------------------------------------------------------------------------
// x_proj: sp[t, 0..159] = hs[t,:] @ x_proj_w[4096,160]. Block per t, 160 thr.
// hs row chunks staged to LDS via TDM (pure copy, no conversion needed).
// ---------------------------------------------------------------------------
__global__ void fm_xproj(const float* __restrict__ hs,
                         const float* __restrict__ xw,
                         float* __restrict__ sp) {
  __shared__ __align__(16) float sh[256];
  int t = blockIdx.x, j = threadIdx.x;
  float acc = 0.0f;
  for (int k0 = 0; k0 < INTER; k0 += 256) {
#ifdef HAVE_TDM
    if (threadIdx.x < 32) {   // one wave issues the DMA
      tdm_copy_f32(sh, hs + (size_t)t * INTER + k0, 256);
      __builtin_amdgcn_s_wait_tensorcnt(0);
    }
#else
    for (int k = threadIdx.x; k < 256; k += blockDim.x)
      sh[k] = hs[(size_t)t * INTER + k0 + k];
#endif
    __syncthreads();
    #pragma unroll 8
    for (int k = 0; k < 256; ++k)
      acc += sh[k] * xw[(size_t)(k0 + k) * (DTRANK + 2 * NSTATE) + j];
    __syncthreads();
  }
  sp[t * (DTRANK + 2 * NSTATE) + j] = acc;
}

// ---------------------------------------------------------------------------
// Weightless RMS over dt(128) / B(16) / C(16).  One wave per timestep.
// ---------------------------------------------------------------------------
__global__ void fm_rms(const float* __restrict__ sp,
                       float* __restrict__ dtn,
                       float* __restrict__ Bn,
                       float* __restrict__ Cn) {
  int t = blockIdx.x, l = threadIdx.x;
  const float* row = sp + t * (DTRANK + 2 * NSTATE);
  float v0 = row[l], v1 = row[l + 32], v2 = row[l + 64], v3 = row[l + 96];
  float ss = v0 * v0 + v1 * v1 + v2 * v2 + v3 * v3;
  #pragma unroll
  for (int o = 16; o > 0; o >>= 1) ss += __shfl_xor(ss, o);
  float sdt = rsqrtf(ss / 128.0f + EPS);
  dtn[t * DTRANK + l]      = v0 * sdt;
  dtn[t * DTRANK + l + 32] = v1 * sdt;
  dtn[t * DTRANK + l + 64] = v2 * sdt;
  dtn[t * DTRANK + l + 96] = v3 * sdt;

  float bv = (l < 16) ? row[DTRANK + l] : 0.0f;
  float sb = bv * bv;
  #pragma unroll
  for (int o = 16; o > 0; o >>= 1) sb += __shfl_xor(sb, o);
  float sB = rsqrtf(sb / 16.0f + EPS);

  float cv = (l < 16) ? row[DTRANK + NSTATE + l] : 0.0f;
  float sc = cv * cv;
  #pragma unroll
  for (int o = 16; o > 0; o >>= 1) sc += __shfl_xor(sc, o);
  float sC = rsqrtf(sc / 16.0f + EPS);

  if (l < 16) {
    Bn[t * NSTATE + l] = bv * sB;
    Cn[t * NSTATE + l] = cv * sC;
  }
}

// ---------------------------------------------------------------------------
// dt = softplus(dt_raw + bias), in place over [SEQ, INTER].
// ---------------------------------------------------------------------------
__global__ void fm_softplus(float* __restrict__ dty, const float* __restrict__ bias) {
  int idx = blockIdx.x * blockDim.x + threadIdx.x;
  int i = idx & (INTER - 1);
  float v = dty[idx] + bias[i];
  dty[idx] = (v > 20.0f) ? v : log1pf(__expf(v));
}

// ---------------------------------------------------------------------------
// Selective scan: thread = channel i, 16-state recurrence in registers.
// B/C chunks staged to LDS via TDM; y written in place over dt (same-thread
// read-then-write, safe).
// ---------------------------------------------------------------------------
#define TCHUNK 64
__global__ __launch_bounds__(128) void fm_scan(
    const float* dt, const float* __restrict__ u,
    const float* __restrict__ Bn, const float* __restrict__ Cn,
    const float* __restrict__ A_log, float* y) {
  __shared__ __align__(16) float sB[TCHUNK * NSTATE];
  __shared__ __align__(16) float sC[TCHUNK * NSTATE];
  int i = blockIdx.x * blockDim.x + threadIdx.x;

  float A[NSTATE], s[NSTATE];
  #pragma unroll
  for (int n = 0; n < NSTATE; ++n) {
    A[n] = -__expf(A_log[(size_t)i * NSTATE + n]);
    s[n] = 0.0f;
  }

  for (int t0 = 0; t0 < SEQ; t0 += TCHUNK) {
#ifdef HAVE_TDM
    if (threadIdx.x < 32) {   // one wave issues both DMAs
      tdm_copy_f32(sB, Bn + (size_t)t0 * NSTATE, TCHUNK * NSTATE);
      tdm_copy_f32(sC, Cn + (size_t)t0 * NSTATE, TCHUNK * NSTATE);
      __builtin_amdgcn_s_wait_tensorcnt(0);
    }
#else
    for (int q = threadIdx.x; q < TCHUNK * NSTATE; q += blockDim.x) {
      sB[q] = Bn[t0 * NSTATE + q];
      sC[q] = Cn[t0 * NSTATE + q];
    }
#endif
    __syncthreads();
    for (int tt = 0; tt < TCHUNK; ++tt) {
      int t = t0 + tt;
      float d  = dt[(size_t)t * INTER + i];
      float uu = u[(size_t)t * INTER + i];
      float du = d * uu;
      float acc = 0.0f;
      #pragma unroll
      for (int n = 0; n < NSTATE; ++n) {
        float dA = __expf(d * A[n]);
        s[n] = dA * s[n] + du * sB[tt * NSTATE + n];
        acc += s[n] * sC[tt * NSTATE + n];
      }
      y[(size_t)t * INTER + i] = acc;
    }
    __syncthreads();
  }
}

// ---------------------------------------------------------------------------
// y2 = (y + u*D) * silu(gate), in place over y.  gate = proj cols [INTER,2I).
// ---------------------------------------------------------------------------
__global__ void fm_gate(float* __restrict__ y, const float* __restrict__ hs,
                        const float* __restrict__ proj, const float* __restrict__ D) {
  int idx = blockIdx.x * blockDim.x + threadIdx.x;
  int i = idx & (INTER - 1);
  int t = idx >> 12;
  float g = proj[(size_t)t * (2 * INTER) + INTER + i];
  float sg = g / (1.0f + __expf(-g));
  y[idx] = (y[idx] + hs[idx] * D[i]) * sg;
}

// ---------------------------------------------------------------------------
extern "C" void kernel_launch(void* const* d_in, const int* in_sizes, int n_in,
                              void* d_out, int out_size, void* d_ws, size_t ws_size,
                              hipStream_t stream) {
  const float* x         = (const float*)d_in[0];
  const float* in_proj_w = (const float*)d_in[1];
  const float* conv_w    = (const float*)d_in[2];
  const float* conv_b    = (const float*)d_in[3];
  const float* x_proj_w  = (const float*)d_in[4];
  const float* dt_proj_w = (const float*)d_in[5];
  const float* dt_proj_b = (const float*)d_in[6];
  const float* out_proj_w= (const float*)d_in[7];
  const float* A_log     = (const float*)d_in[8];
  const float* D_param   = (const float*)d_in[9];
  float* out = (float*)d_out;

  float* ws   = (float*)d_ws;
  float* proj = ws;                                   // SEQ*2I   = 8,388,608
  float* hs   = proj + (size_t)SEQ * 2 * INTER;       // SEQ*I    = 4,194,304
  float* sp   = hs   + (size_t)SEQ * INTER;           // SEQ*160
  float* dtn  = sp   + (size_t)SEQ * (DTRANK + 2 * NSTATE);
  float* Bn   = dtn  + (size_t)SEQ * DTRANK;
  float* Cn   = Bn   + (size_t)SEQ * NSTATE;
  float* dty  = Cn   + (size_t)SEQ * NSTATE;          // dt -> y -> y2 (in place)

  // 1) proj = x @ in_proj_w                 [1024,2048] x [2048,8192]
  fm_gemm_bf16_wmma<<<dim3((2 * INTER) / BN, SEQ / BM), 256, 0, stream>>>(
      x, in_proj_w, proj, SEQ, 2 * INTER, HID);
  // 2) causal depthwise conv + SiLU
  fm_conv_silu<<<(SEQ * INTER) / 256, 256, 0, stream>>>(proj, conv_w, conv_b, hs);
  // 3) ssm_p = hs @ x_proj_w                [1024,4096] x [4096,160]
  fm_xproj<<<SEQ, DTRANK + 2 * NSTATE, 0, stream>>>(hs, x_proj_w, sp);
  // 4) RMS norms of dt / B / C
  fm_rms<<<SEQ, 32, 0, stream>>>(sp, dtn, Bn, Cn);
  // 5) dt_raw = dtn @ dt_proj_w             [1024,128] x [128,4096]
  fm_gemm_bf16_wmma<<<dim3(INTER / BN, SEQ / BM), 256, 0, stream>>>(
      dtn, dt_proj_w, dty, SEQ, INTER, DTRANK);
  // 6) dt = softplus(dt_raw + b)
  fm_softplus<<<(SEQ * INTER) / 256, 256, 0, stream>>>(dty, dt_proj_b);
  // 7) selective scan (y over dt, in place)
  fm_scan<<<INTER / 128, 128, 0, stream>>>(dty, hs, Bn, Cn, A_log, dty);
  // 8) y2 = (y + u*D) * silu(gate)
  fm_gate<<<(SEQ * INTER) / 256, 256, 0, stream>>>(dty, hs, proj, D_param);
  // 9) out = y2 @ out_proj_w                [1024,4096] x [4096,2048]
  fm_gemm_bf16_wmma<<<dim3(HID / BN, SEQ / BM), 256, 0, stream>>>(
      dty, out_proj_w, out, SEQ, HID, INTER);
}
